// Global_interaction_mult_27341761806364
// MI455X (gfx1250) — matrix operand order repaired and checked by reading, not compiled, not run on hardware
//
#include <hip/hip_runtime.h>

typedef __attribute__((ext_vector_type(16))) _Float16 v16h;
typedef __attribute__((ext_vector_type(8)))  float    v8f;

constexpr int S = 20, N = 256, D = 64;
constexpr int RPAD = 72;   // _Float16 row stride for R / WgA (pad vs 64 banks)
constexpr int LPAD = 65;   // float row stride for Lin buffer

// ---------------------------------------------------------------------------
// Kernel 1: per-node terms  hWgB = h@Wg[:,D:2D]^T, hWgC = h@Wg[:,2D:]^T,
//           hWaB = h@Wa[D:2D], hWaC = h@Wa[2D:]   (avoids N-fold recompute)
// ---------------------------------------------------------------------------
__global__ __launch_bounds__(256)
void precompute_h_terms(const float* __restrict__ hidden,
                        const float* __restrict__ Wg,
                        const float* __restrict__ Wa,
                        float* __restrict__ hWgB, float* __restrict__ hWgC,
                        float* __restrict__ hWaB, float* __restrict__ hWaC)
{
    __shared__ float sh[4][64];
    const int r   = threadIdx.x >> 6;
    const int d   = threadIdx.x & 63;
    const int row = blockIdx.x * 4 + r;
    sh[r][d] = hidden[row * 64 + d];
    __syncthreads();
    float aB = 0.f, aC = 0.f;
    #pragma unroll 8
    for (int k = 0; k < 64; ++k) {
        const float h = sh[r][k];
        aB += h * Wg[d * 192 + 64  + k];
        aC += h * Wg[d * 192 + 128 + k];
    }
    hWgB[row * 64 + d] = aB;
    hWgC[row * 64 + d] = aC;
    if (d == 0) {
        float b = 0.f, c = 0.f;
        for (int k = 0; k < 64; ++k) {
            b += sh[r][k] * Wa[64  + k];
            c += sh[r][k] * Wa[128 + k];
        }
        hWaB[row] = b;
        hWaC[row] = c;
    }
}

// ---------------------------------------------------------------------------
// Kernel 2: fully fused per-(s,i) block. 256 threads = 8 wave32.
// ---------------------------------------------------------------------------
struct alignas(16) Smem {
    float    sWr[64 * 6];
    float    sbr[64], sgr[64], sbetar[64];
    float    sbg[64], sgg[64], sbetag[64];
    float    swa[64];
    float    sHi[64];                 // hWgB row for node i
    float    sLin[256 * LPAD];        // lin_r -> LinG -> contributions (reused)
    float    red0[256], red1[256];
    float    sHin[64];
    float    sc[4];                   // [0]=hWaB_i+ba, [1]=u3, [2]=rstd3
    _Float16 sWgA[64 * RPAD];         // Wg[:, :D] as f16, row d, 64 k's
    _Float16 sR[256 * RPAD];          // r in f16, row j, 64 k's
};

__global__ __launch_bounds__(256)
void fused_interaction(const float* __restrict__ corr_index,
                       const int*   __restrict__ nei_index,
                       const float* __restrict__ hidden,
                       const float* __restrict__ dest_corr,
                       const float* __restrict__ agent_v,
                       const float* __restrict__ Wr,  const float* __restrict__ br,
                       const float* __restrict__ gr,  const float* __restrict__ betar,
                       const float* __restrict__ Wg,  const float* __restrict__ bg_,
                       const float* __restrict__ gg,  const float* __restrict__ betag,
                       const float* __restrict__ Wa,  const float* __restrict__ ba,
                       const float* __restrict__ Ww,  const float* __restrict__ bw,
                       const float* __restrict__ gw,  const float* __restrict__ betaw,
                       const float* __restrict__ hWgB, const float* __restrict__ hWgC,
                       const float* __restrict__ hWaB, const float* __restrict__ hWaC,
                       float* __restrict__ out)
{
    __shared__ Smem sm;
    const int blk  = blockIdx.x;
    const int s    = blk / N;
    const int i    = blk % N;
    const int tid  = threadIdx.x;
    const int lane = tid & 31;
    const int wave = tid >> 5;

    // ---- stage weights ----
    for (int idx = tid; idx < 64 * 6; idx += 256) sm.sWr[idx] = Wr[idx];
    for (int idx = tid; idx < 64 * 64; idx += 256) {
        const int d = idx >> 6, k = idx & 63;
        sm.sWgA[d * RPAD + k] = (_Float16)Wg[d * 192 + k];
    }
    if (tid < 64) {
        sm.sbr[tid] = br[tid];  sm.sgr[tid] = gr[tid];  sm.sbetar[tid] = betar[tid];
        sm.sbg[tid] = bg_[tid]; sm.sgg[tid] = gg[tid];  sm.sbetag[tid] = betag[tid];
        sm.swa[tid] = Wa[tid];
        sm.sHi[tid] = hWgB[(size_t)(s * N + i) * 64 + tid];
    }
    if (tid == 0) sm.sc[0] = hWaB[s * N + i] + ba[0];
    __syncthreads();

    // ---- Phase A: per-pair feature MLP (K=6) + LN + relu -> r (f16), tt ----
    const int j = tid;
    const float c0 = corr_index[(i * N + j) * 2 + 0];
    const float c1 = corr_index[(i * N + j) * 2 + 1];
    const float c2 = dest_corr[((size_t)(s * N + i) * N + j) * 2 + 0];
    const float c3 = dest_corr[((size_t)(s * N + i) * N + j) * 2 + 1];
    const float c4 = agent_v[j * 2 + 0];
    const float c5 = agent_v[j * 2 + 1];
    float* lrow = &sm.sLin[j * LPAD];
    float sum = 0.f, sq = 0.f;
    #pragma unroll 4
    for (int d = 0; d < 64; ++d) {
        const float* w = &sm.sWr[d * 6];
        float v = sm.sbr[d] + w[0]*c0 + w[1]*c1 + w[2]*c2 + w[3]*c3 + w[4]*c4 + w[5]*c5;
        lrow[d] = v; sum += v; sq += v * v;
    }
    float u    = sum * (1.f / 64.f);
    float rstd = rsqrtf(sq * (1.f / 64.f) - u * u + 1e-5f);
    _Float16* rrow = &sm.sR[j * RPAD];
    float tt = 0.f;
    #pragma unroll 4
    for (int d = 0; d < 64; ++d) {
        float rv = (lrow[d] - u) * rstd * sm.sgr[d] + sm.sbetar[d];
        rv = fmaxf(rv, 0.f);
        rrow[d] = (_Float16)rv;
        tt += rv * sm.swa[d];
    }
    tt = fmaxf(tt + sm.sc[0] + hWaC[s * N + j], 0.f);
    const bool msk = nei_index[i * N + j] > 0;
    float Pos = msk ? tt : 0.f;
    if (Pos == 0.f) Pos = -10000.f;

    // ---- softmax over j ----
    sm.red0[j] = Pos;
    __syncthreads();
    for (int off = 128; off > 0; off >>= 1) {
        if (tid < off) sm.red0[tid] = fmaxf(sm.red0[tid], sm.red0[tid + off]);
        __syncthreads();
    }
    const float mx = sm.red0[0];
    __syncthreads();
    const float e = __expf(Pos - mx);
    sm.red1[j] = e;
    __syncthreads();
    for (int off = 128; off > 0; off >>= 1) {
        if (tid < off) sm.red1[tid] += sm.red1[tid + off];
        __syncthreads();
    }
    const float pj = e / sm.red1[0];
    __syncthreads();

    // ---- Phase B: LinG = R(256x64,f16) x WgA^T(64x64,f16) via v_wmma ----
    // A frag (16x32 f16): lane m=lane&15; VGPR v: k = ks*32 + 2v + (lane>=16?8:0) + (v>=4?8:0)
    // B frag (32x16 f16): lane n=lane&15; VGPR v: k = ks*32 + 2v + (lane>=16?8:0) + (v>=4?8:0)
    for (int jt = wave; jt < 16; jt += 8) {
        const int row = jt * 16 + (lane & 15);
        const _Float16* ar = &sm.sR[row * RPAD];
        for (int nt = 0; nt < 4; ++nt) {
            const int dcol = nt * 16 + (lane & 15);
            const _Float16* brw = &sm.sWgA[dcol * RPAD];
            v8f acc = {};
            #pragma unroll
            for (int ks = 0; ks < 2; ++ks) {
                union { unsigned int u[8]; v16h h; } Af, Bf;
                #pragma unroll
                for (int vv = 0; vv < 8; ++vv) {
                    const int ka = ks * 32 + 2 * vv + ((lane & 16) ? 8 : 0) + ((vv & 4) ? 8 : 0);
                    const int kb = ks * 32 + 2 * vv + ((lane & 16) ? 8 : 0) + ((vv & 4) ? 8 : 0);
                    Af.u[vv] = *(const unsigned int*)&ar[ka];
                    Bf.u[vv] = *(const unsigned int*)&brw[kb];
                }
                acc = __builtin_amdgcn_wmma_f32_16x16x32_f16(
                          false, Af.h, false, Bf.h, (short)0, acc, false, false);
            }
            #pragma unroll
            for (int vv = 0; vv < 8; ++vv) {
                const int jj = jt * 16 + vv + ((lane & 16) ? 8 : 0);
                sm.sLin[jj * LPAD + nt * 16 + (lane & 15)] = acc[vv];
            }
        }
    }
    __syncthreads();

    // ---- Phase C: gate LN+sigmoid, weighted contribution ----
    const float* hc = &hWgC[(size_t)(s * N + j) * 64];
    float sum2 = 0.f, sq2 = 0.f;
    #pragma unroll 4
    for (int d = 0; d < 64; ++d) {
        float v = lrow[d] + sm.sHi[d] + hc[d] + sm.sbg[d];
        lrow[d] = v; sum2 += v; sq2 += v * v;
    }
    const float u2    = sum2 * (1.f / 64.f);
    const float rstd2 = rsqrtf(sq2 * (1.f / 64.f) - u2 * u2 + 1e-5f);
    const float wj    = msk ? pj : 0.f;
    const float* hj   = &hidden[(size_t)(s * N + j) * 64];
    #pragma unroll 4
    for (int d = 0; d < 64; ++d) {
        const float x = (lrow[d] - u2) * rstd2 * sm.sgg[d] + sm.sbetag[d];
        const float g = 1.f / (1.f + __expf(-x));
        lrow[d] = wj * g * hj[d];
    }
    __syncthreads();

    // ---- reduce over j: H_in[d] = sum_j contrib[j][d] ----
    {
        const int d = tid & 63, grp = tid >> 6;
        float acc = 0.f;
        for (int jj = grp * 64; jj < grp * 64 + 64; ++jj) acc += sm.sLin[jj * LPAD + d];
        sm.red0[grp * 64 + d] = acc;
    }
    __syncthreads();
    if (tid < 64)
        sm.sHin[tid] = sm.red0[tid] + sm.red0[64 + tid] + sm.red0[128 + tid] + sm.red0[192 + tid];
    __syncthreads();

    // ---- final matvec + LN + relu + residual ----
    if (tid < 64) {
        float v = bw[tid];
        #pragma unroll 8
        for (int k = 0; k < 64; ++k) v += sm.sHin[k] * Ww[tid * 64 + k];
        sm.red1[tid] = v;
    }
    __syncthreads();
    if (tid == 0) {
        float su = 0.f, ss = 0.f;
        for (int d = 0; d < 64; ++d) { const float v = sm.red1[d]; su += v; ss += v * v; }
        const float u3 = su * (1.f / 64.f);
        sm.sc[1] = u3;
        sm.sc[2] = rsqrtf(ss * (1.f / 64.f) - u3 * u3 + 1e-5f);
    }
    __syncthreads();
    if (tid < 64) {
        const float hs = fmaxf((sm.red1[tid] - sm.sc[1]) * sm.sc[2] * gw[tid] + betaw[tid], 0.f);
        out[(size_t)(s * N + i) * 64 + tid] = hidden[(size_t)(s * N + i) * 64 + tid] + hs;
    }
}

// ---------------------------------------------------------------------------
extern "C" void kernel_launch(void* const* d_in, const int* in_sizes, int n_in,
                              void* d_out, int out_size, void* d_ws, size_t ws_size,
                              hipStream_t stream)
{
    const float* corr_index = (const float*)d_in[0];
    const int*   nei_index  = (const int*)  d_in[1];
    // d_in[2] nei_num  : unused by reference
    const float* hidden     = (const float*)d_in[3];
    const float* dest_corr  = (const float*)d_in[4];
    // d_in[5] past_dest: unused by reference
    const float* agent_v    = (const float*)d_in[6];
    const float* Wr    = (const float*)d_in[7];
    const float* br    = (const float*)d_in[8];
    const float* gr    = (const float*)d_in[9];
    const float* betar = (const float*)d_in[10];
    const float* Wg    = (const float*)d_in[11];
    const float* bg    = (const float*)d_in[12];
    const float* gg    = (const float*)d_in[13];
    const float* betag = (const float*)d_in[14];
    const float* Wa    = (const float*)d_in[15];
    const float* ba    = (const float*)d_in[16];
    const float* Ww    = (const float*)d_in[17];
    const float* bw    = (const float*)d_in[18];
    const float* gw    = (const float*)d_in[19];
    const float* betaw = (const float*)d_in[20];
    float* out = (float*)d_out;

    // workspace layout (floats): hWgB[S*N*64] | hWgC[S*N*64] | hWaB[S*N] | hWaC[S*N]
    float* hWgB = (float*)d_ws;
    float* hWgC = hWgB + (size_t)S * N * 64;
    float* hWaB = hWgC + (size_t)S * N * 64;
    float* hWaC = hWaB + (size_t)S * N;

    precompute_h_terms<<<(S * N) / 4, 256, 0, stream>>>(hidden, Wg, Wa, hWgB, hWgC, hWaB, hWaC);
    fused_interaction<<<S * N, 256, 0, stream>>>(
        corr_index, nei_index, hidden, dest_corr, agent_v,
        Wr, br, gr, betar, Wg, bg, gg, betag, Wa, ba, Ww, bw, gw, betaw,
        hWgB, hWgC, hWaB, hWaC, out);
}
